// HankelFilterBank_17059610100042
// MI455X (gfx1250) — matrix operand order repaired and checked by reading, not compiled, not run on hardware
//
#include <hip/hip_runtime.h>
#include <hip/hip_bf16.h>

typedef __attribute__((ext_vector_type(16))) _Float16 v16h;
typedef __attribute__((ext_vector_type(8)))  float    v8f;

#define B_    16
#define L_    65536
#define H_    64
#define K_    64
#define WARM  (H_ - 1)
#define LOUT  (L_ - WARM)      // 65473
#define TROWS 128              // output rows per block
#define NWAVE 8                // 256 threads, wave32
#define OSTR  68               // padded LDS row stride (dwords) for D staging

__global__ __launch_bounds__(256)
void hankel_fb_kernel(const float* __restrict__ x,
                      const float* __restrict__ W,
                      const float* __restrict__ bias,
                      float* __restrict__ out)
{
    __shared__ _Float16 Wlds[K_ * H_];                 // W[k][h] as f16 (8 KB)
    __shared__ float    xs[TROWS + H_];                // x chunk: 191 used
    __shared__ float    muL[TROWS];
    __shared__ float    scL[TROWS];
    __shared__ __align__(16) float outT[NWAVE][16][OSTR];  // D staging (~34.8 KB)

    const int tid  = threadIdx.x;
    const int wave = tid >> 5;
    const int lane = tid & 31;
    const int m    = lane & 15;   // row/col within 16-wide half
    const int khi  = lane >> 4;   // K-half selector per WMMA layout

    const int b  = blockIdx.y;
    const int t0 = blockIdx.x * TROWS;

    // ---- stage W (f32 -> f16) into LDS ----
    for (int i = tid; i < K_ * H_; i += 256)
        Wlds[i] = (_Float16)W[i];

    // ---- stage x window [t0, t0 + TROWS + WARM) ----
    const float* xb = x + (size_t)b * L_;
    for (int i = tid; i < TROWS + WARM; i += 256) {
        int gi = t0 + i;
        xs[i] = (gi < L_) ? xb[gi] : 0.0f;
    }
    // hint: pull next tile's chunk toward the caches (global_prefetch_b8)
    if (t0 + TROWS < L_)
        __builtin_prefetch(xb + t0 + TROWS + tid, 0, 1);
    __syncthreads();

    // ---- per-row mean / inv-std (ddof=1), threads 0..127 each own a row ----
    if (tid < TROWS) {
        float s = 0.f, sq = 0.f;
        #pragma unroll
        for (int j = 0; j < H_; ++j) {
            float v = xs[tid + j];
            s += v; sq += v * v;
        }
        float mu  = s * (1.0f / H_);
        float var = (sq - s * mu) * (1.0f / (H_ - 1));
        var = fmaxf(var, 0.0f);
        muL[tid] = mu;
        scL[tid] = 1.0f / (sqrtf(var) + 1e-6f);
    }
    __syncthreads();

    // ---- build A fragments: 16x32 f16 (ISA 7.12.2 A-layout) ----
    // lane m holds row M=m; lanes 16-31 hold the K+8 block;
    // VGPR v, half h: K = (v>=4 ? 16 : 0) + 8*khi + (v&3)*2 + h
    const int row = wave * 16 + m;          // local row in [0,128)
    const float mu = muL[row];
    const float sc = scL[row];
    v16h a0, a1;
    #pragma unroll
    for (int e = 0; e < 16; ++e) {
        int v = e >> 1, h = e & 1;
        int Kk = ((v >= 4) ? 16 : 0) + 8 * khi + (v & 3) * 2 + h;
        a0[e] = (_Float16)((xs[row + Kk]      - mu) * sc);
        a1[e] = (_Float16)((xs[row + Kk + 32] - mu) * sc);
    }

    // ---- gather B fragments from Wlds: B[h][n] = W[n][h] ----
    // B layout: lane l holds col N=l%16; element e: K = (e>=8 ? 16 : 0) + 8*(l/16) + (e&7)
    v16h bf[2][4];
    #pragma unroll
    for (int kc = 0; kc < 2; ++kc) {
        #pragma unroll
        for (int nt = 0; nt < 4; ++nt) {
            int N = nt * 16 + m;
            #pragma unroll
            for (int e = 0; e < 16; ++e) {
                int Kh = kc * 32 + ((e >= 8) ? 16 : 0) + 8 * khi + (e & 7);
                bf[kc][nt][e] = Wlds[N * H_ + Kh];
            }
        }
    }

    // ---- 4 N-tiles x (K=64 as 2 x 16x16x32 WMMA); stage D into LDS ----
    #pragma unroll
    for (int nt = 0; nt < 4; ++nt) {
        v8f c = {};
        c = __builtin_amdgcn_wmma_f32_16x16x32_f16(false, a0, false, bf[0][nt],
                                                   (short)0, c, false, false);
        c = __builtin_amdgcn_wmma_f32_16x16x32_f16(false, a1, false, bf[1][nt],
                                                   (short)0, c, false, false);
        int   kcol = nt * 16 + m;
        float bb   = bias[kcol];
        #pragma unroll
        for (int r = 0; r < 8; ++r)   // D layout: VGPR r -> M = r + 8*khi
            outT[wave][r + 8 * khi][kcol] = fmaxf(c[r] + bb, 0.0f);
    }
    // intra-wave LDS RAW: hardware/DScnt ordering, no block barrier needed

    // ---- coalesced b128 stores: 2 contiguous 256B rows per instruction ----
    float* outb = out + (size_t)b * LOUT * K_;
    const int trow0 = t0 + wave * 16;
    const int col   = 4 * m;
    if (t0 + TROWS <= LOUT) {           // block-uniform fast path (511/512)
        #pragma unroll
        for (int j = 0; j < 8; ++j) {
            int r = 2 * j + khi;
            float4 v4 = *(const float4*)&outT[wave][r][col];
            *(float4*)(outb + (size_t)(trow0 + r) * K_ + col) = v4;
        }
    } else {                            // ragged last tile only
        #pragma unroll
        for (int j = 0; j < 8; ++j) {
            int r = 2 * j + khi;
            int t = trow0 + r;
            if (t < LOUT) {
                float4 v4 = *(const float4*)&outT[wave][r][col];
                *(float4*)(outb + (size_t)t * K_ + col) = v4;
            }
        }
    }
}

extern "C" void kernel_launch(void* const* d_in, const int* in_sizes, int n_in,
                              void* d_out, int out_size, void* d_ws, size_t ws_size,
                              hipStream_t stream) {
    (void)in_sizes; (void)n_in; (void)d_ws; (void)ws_size; (void)out_size;
    const float* x    = (const float*)d_in[0];   // (16, 65536) f32
    const float* W    = (const float*)d_in[1];   // (64, 64)    f32
    const float* bias = (const float*)d_in[2];   // (64,)       f32
    float*       out  = (float*)d_out;           // (16, 65473, 64) f32

    dim3 grid((LOUT + TROWS - 1) / TROWS, B_);   // (512, 16)
    dim3 block(256);
    hankel_fb_kernel<<<grid, block, 0, stream>>>(x, W, bias, out);
}
